// ISTFTHead_19327352832077
// MI455X (gfx1250) — compile-verified, matrix operand order checked by reading, not run on hardware
//
#include <hip/hip_runtime.h>
#include <hip/hip_bf16.h>
#include <math.h>

// ---------------------------------------------------------------------------
// ISTFT head for MI455X (gfx1250, wave32, WMMA bf16 16x16x32).
//   Stage 1: cast/prepare bf16 operands + windowed-DFT basis.
//   Stage 2: GEMM1 (x @ W^T) with fused exp/clamp/cos/sin epilogue -> A matrix.
//           W mag+phase tiles (shared by all 4 waves) staged in LDS via
//           double-buffered GLOBAL_LOAD_ASYNC_TO_LDS_B128.
//   Stage 3: GEMM2 (A @ basis) = irfft+window, epilogue = overlap-add atomics.
//           A tile (shared by all 4 waves) staged via async-to-LDS B64.
//   Stage 4: envelope-normalize + crop.
// ---------------------------------------------------------------------------

typedef __attribute__((ext_vector_type(16))) __bf16 v16bf;
typedef __attribute__((ext_vector_type(8)))  float  v8f;

namespace {
constexpr int kBatch  = 16;
constexpr int kT      = 4096;          // frames (= L)
constexpr int kDim    = 512;
constexpr int kOut    = 1026;          // 2*513
constexpr int kNfft   = 1024;
constexpr int kHop    = 256;
constexpr int kPad    = 384;
constexpr int kRows   = kBatch * kT;   // 65536 GEMM rows
constexpr int kK2     = 1056;          // spectral K padded to 33*32
constexpr int kYlen   = (kT - 1) * kHop + kNfft;   // 1049344
constexpr int kOutLen = 1 << 20;       // per-batch output samples

constexpr size_t OFF_XB = 0;
constexpr size_t SZ_XB  = (size_t)kRows * kDim * 2;     // x in bf16
constexpr size_t OFF_WB = OFF_XB + SZ_XB;
constexpr size_t SZ_WB  = (size_t)kK2 * kDim * 2;       // W bf16, zero-padded rows
constexpr size_t OFF_A2 = OFF_WB + SZ_WB;
constexpr size_t SZ_A2  = (size_t)kRows * kK2 * 2;      // activated spectrum (bf16)
constexpr size_t OFF_B2 = OFF_A2 + SZ_A2;
constexpr size_t SZ_B2  = (size_t)kNfft * kK2 * 2;      // basis^T (bf16)
constexpr size_t OFF_Y  = OFF_B2 + SZ_B2;
constexpr size_t SZ_Y   = (size_t)kBatch * kYlen * 4;   // OLA accumulator (f32)
constexpr float  kTwoPiOverN = 6.283185307179586f / 1024.0f;
}

__device__ __forceinline__ unsigned short f2b(float f) {
  unsigned int u = __float_as_uint(f);
  unsigned int r = u + 0x7FFFu + ((u >> 16) & 1u);   // round-to-nearest-even
  return (unsigned short)(r >> 16);
}

union Frag { uint4 q[2]; v16bf v; };

// LDS offset of a __shared__ object: flat LDS pointers carry the LDS byte
// offset in their low 32 bits (aperture bits live in [63:32]).
__device__ __forceinline__ unsigned lds_off(const void* p) {
  return (unsigned)(size_t)p;
}

// Async global->LDS copies (ASYNCcnt-tracked, overlap with WMMA).
__device__ __forceinline__ void async_g2l_b128(unsigned lds, const void* g) {
  asm volatile("global_load_async_to_lds_b128 %0, %1, off"
               :: "v"(lds), "v"(g) : "memory");
}
__device__ __forceinline__ void async_g2l_b64(unsigned lds, const void* g) {
  asm volatile("global_load_async_to_lds_b64 %0, %1, off"
               :: "v"(lds), "v"(g) : "memory");
}
__device__ __forceinline__ void wait_async0() {
  asm volatile("s_wait_asynccnt 0x0" ::: "memory");
}

// ---------------- stage 1: operand preparation ----------------------------

__global__ void k_cast_x(const float* __restrict__ x, unsigned short* __restrict__ xb, int n) {
  int i = blockIdx.x * blockDim.x + threadIdx.x;
  if (i < n) xb[i] = f2b(x[i]);
}

__global__ void k_prep_w(const float* __restrict__ W, unsigned short* __restrict__ wb) {
  int i = blockIdx.x * blockDim.x + threadIdx.x;        // over kK2 * kDim
  if (i >= kK2 * kDim) return;
  int o = i >> 9, k = i & 511;
  wb[i] = (o < kOut) ? f2b(W[o * kDim + k]) : (unsigned short)0;
}

// basisT[n][col]: col<513 -> w(n)*c_f*cos(2*pi*f*n/N); col in [513,1026) ->
// w(n)*c_f*sin(...); col>=1026 -> 0. Window + 1/N scaling folded in.
__global__ void k_basis(unsigned short* __restrict__ b2t) {
  int col = blockIdx.x * blockDim.x + threadIdx.x;
  int n   = blockIdx.y;
  if (col >= kK2) return;
  float val = 0.0f;
  if (col < kOut) {
    int f = (col < 513) ? col : col - 513;
    float c  = ((f == 0) || (f == 512)) ? (1.0f / 1024.0f) : (2.0f / 1024.0f);
    float wn = 0.5f - 0.5f * __cosf(kTwoPiOverN * (float)n);
    int   p  = (f * n) & 1023;                     // exact range reduction
    float a  = kTwoPiOverN * (float)p;
    val = wn * c * ((col < 513) ? __cosf(a) : __sinf(a));
  }
  b2t[(size_t)n * kK2 + col] = f2b(val);
}

__global__ void k_zero_padA(unsigned short* __restrict__ a2) {
  int r = blockIdx.x * blockDim.x + threadIdx.x;   // one row each
  if (r >= kRows) return;
  unsigned int* p = (unsigned int*)(a2 + (size_t)r * kK2 + kOut);  // cols 1026..1055
  #pragma unroll
  for (int j = 0; j < 15; ++j) p[j] = 0u;
}

// ---------------- stage 2: GEMM1 + activation epilogue --------------------
// Block = 4 waves * 16 rows, one 16-wide frequency tile o0. The W mag tile
// (rows o0..o0+15) and phase tile (rows o0+513..) are identical for all 4
// waves -> staged once per block in LDS with double-buffered async copies.

__global__ void __launch_bounds__(128)
k_gemm1_act(const unsigned short* __restrict__ xb,
            const unsigned short* __restrict__ wb,
            const float* __restrict__ bias,
            unsigned short* __restrict__ a2) {
  __shared__ __align__(16) unsigned short ldsW[2][2 * 16 * 32]; // [buf][half*512+r*32+k]

  const int wid  = threadIdx.x >> 5;
  const int lane = threadIdx.x & 31;
  const int lm   = lane & 15;
  const int hi   = lane >> 4;
  const int koff = hi * 8;                 // lo lanes: K 0-7/16-23, hi: 8-15/24-31
  const int row0 = (blockIdx.x * 4 + wid) * 16;
  const int o0   = blockIdx.y * 16;

  const unsigned short* xr = xb + (size_t)(row0 + lm) * kDim;
  const unsigned ldsWb = lds_off(&ldsW[0][0]);

  // 128 threads stage 2 KB (mag 16x32 + phase 16x32) per k-slice, 16B each.
  auto stage_w = [&](int k0, int buf) {
    const int i = threadIdx.x;
    const int h = i >> 6;                  // 0 = mag rows, 1 = phase rows
    const int r = (i >> 2) & 15;
    const int c = i & 3;                   // 16-byte chunk within the row
    const unsigned short* g = wb + (size_t)(o0 + h * 513 + r) * kDim + k0 + c * 8;
    async_g2l_b128(ldsWb + (unsigned)(buf * 2048 + h * 1024 + r * 64 + c * 16), g);
  };

  v8f accm = {0.f,0.f,0.f,0.f,0.f,0.f,0.f,0.f};
  v8f accp = {0.f,0.f,0.f,0.f,0.f,0.f,0.f,0.f};

  int buf = 0;
  stage_w(0, 0);
  wait_async0();
  __syncthreads();

  constexpr int kSteps = kDim / 32;        // 16
  for (int it = 0; it < kSteps; ++it) {
    const int k0 = it * 32;
    if (it + 1 < kSteps) stage_w(k0 + 32, buf ^ 1);
    __builtin_prefetch(xr + k0 + 128, 0, 3);

    const unsigned short* lw = &ldsW[buf][0];
    Frag a, bm, bp;
    a.q[0]  = *(const uint4*)(xr + k0 + koff);
    a.q[1]  = *(const uint4*)(xr + k0 + koff + 16);
    bm.q[0] = *(const uint4*)(lw + lm * 32 + koff);          // ds_load_b128
    bm.q[1] = *(const uint4*)(lw + lm * 32 + koff + 16);
    bp.q[0] = *(const uint4*)(lw + 512 + lm * 32 + koff);
    bp.q[1] = *(const uint4*)(lw + 512 + lm * 32 + koff + 16);
    accm = __builtin_amdgcn_wmma_f32_16x16x32_bf16(false, a.v, false, bm.v,
                                                   (short)0, accm, false, false);
    accp = __builtin_amdgcn_wmma_f32_16x16x32_bf16(false, a.v, false, bp.v,
                                                   (short)0, accp, false, false);
    if (it + 1 < kSteps) {
      wait_async0();
      __syncthreads();
      buf ^= 1;
    }
  }

  const int f = o0 + lm;                    // frequency bin handled by this lane
  if (f <= 512) {
    const float bm = bias[f];
    const float bp = bias[513 + f];
    unsigned short* areal = a2 + f;
    unsigned short* aimag = a2 + 513 + f;
    #pragma unroll
    for (int j = 0; j < 8; ++j) {
      const int r = row0 + j + hi * 8;
      float m  = fminf(__expf(accm[j] + bm), 100.0f);
      float ph = accp[j] + bp;
      areal[(size_t)r * kK2] = f2b(m * __cosf(ph));
      aimag[(size_t)r * kK2] = f2b(-m * __sinf(ph));    // minus sign folded here
    }
  }
}

// ---------------- stage 3: GEMM2 (irfft+window) + overlap-add -------------
// Block = one 16-row frame tile (shared by all 4 waves) x 4 sample tiles.
// The shared A tile slice is staged once per block via async-to-LDS.

__global__ void __launch_bounds__(128)
k_gemm2_ola(const unsigned short* __restrict__ a2,
            const unsigned short* __restrict__ b2t,
            float* __restrict__ y) {
  __shared__ __align__(16) unsigned short ldsA[2][16 * 32];   // [buf][r*32+k]

  const int wid  = threadIdx.x >> 5;
  const int lane = threadIdx.x & 31;
  const int lm   = lane & 15;
  const int hi   = lane >> 4;
  const int koff = hi * 8;
  const int row0 = blockIdx.x * 16;                 // frame tile (shared by 4 waves)
  const int n0   = (blockIdx.y * 4 + wid) * 16;     // sample offset within frame

  const unsigned short* br = b2t + (size_t)(n0 + lm) * kK2;
  const unsigned ldsAb = lds_off(&ldsA[0][0]);

  // 128 threads stage the 1 KB A slice (16 rows x 32 k), 8 bytes each.
  auto stage_a = [&](int k0, int buf) {
    const int i = threadIdx.x;
    const int r = i >> 3;
    const int c = i & 7;                   // 8-byte chunk within the row
    const unsigned short* g = a2 + (size_t)(row0 + r) * kK2 + k0 + c * 4;
    async_g2l_b64(ldsAb + (unsigned)(buf * 1024 + r * 64 + c * 8), g);
  };

  v8f acc = {0.f,0.f,0.f,0.f,0.f,0.f,0.f,0.f};

  int buf = 0;
  stage_a(0, 0);
  wait_async0();
  __syncthreads();

  constexpr int kSteps = kK2 / 32;         // 33
  for (int it = 0; it < kSteps; ++it) {
    const int k0 = it * 32;
    if (it + 1 < kSteps) stage_a(k0 + 32, buf ^ 1);
    __builtin_prefetch(br + k0 + 128, 0, 3);

    const unsigned short* la = &ldsA[buf][0];
    Frag a, b;
    a.q[0] = *(const uint4*)(la + lm * 32 + koff);           // ds_load_b128
    a.q[1] = *(const uint4*)(la + lm * 32 + koff + 16);
    b.q[0] = *(const uint4*)(br + k0 + koff);
    b.q[1] = *(const uint4*)(br + k0 + koff + 16);
    acc = __builtin_amdgcn_wmma_f32_16x16x32_bf16(false, a.v, false, b.v,
                                                  (short)0, acc, false, false);
    if (it + 1 < kSteps) {
      wait_async0();
      __syncthreads();
      buf ^= 1;
    }
  }

  const int n = n0 + lm;
  #pragma unroll
  for (int j = 0; j < 8; ++j) {
    const int r = row0 + j + hi * 8;
    const int b = r >> 12;            // batch (4096 frames per batch)
    const int t = r & 4095;           // frame index
    atomicAdd(&y[(size_t)b * kYlen + t * kHop + n], acc[j]);
  }
}

// ---------------- stage 4: envelope normalize + crop ----------------------

__global__ void k_finalize(const float* __restrict__ y, float* __restrict__ out) {
  int idx = blockIdx.x * blockDim.x + threadIdx.x;
  if (idx >= kBatch * kOutLen) return;
  const int b = idx >> 20;
  const int i = idx & (kOutLen - 1);
  const int m = i + kPad;
  int tmin = (m - 768) >> 8;  if (tmin < 0) tmin = 0;        // ceil((m-1023)/256)
  int tmax = m >> 8;          if (tmax > kT - 1) tmax = kT - 1;
  float env = 0.0f;
  #pragma unroll 4
  for (int t = tmin; t <= tmax; ++t) {
    const float w = 0.5f - 0.5f * __cosf(kTwoPiOverN * (float)(m - t * kHop));
    env += w * w;
  }
  out[idx] = y[(size_t)b * kYlen + m] / env;
}

// ---------------------------------------------------------------------------

extern "C" void kernel_launch(void* const* d_in, const int* in_sizes, int n_in,
                              void* d_out, int out_size, void* d_ws, size_t ws_size,
                              hipStream_t stream) {
  const float* x    = (const float*)d_in[0];
  const float* W    = (const float*)d_in[1];
  const float* bias = (const float*)d_in[2];
  float* out = (float*)d_out;

  char* ws = (char*)d_ws;
  unsigned short* XB  = (unsigned short*)(ws + OFF_XB);
  unsigned short* WB  = (unsigned short*)(ws + OFF_WB);
  unsigned short* A2  = (unsigned short*)(ws + OFF_A2);
  unsigned short* B2T = (unsigned short*)(ws + OFF_B2);
  float*          Y   = (float*)(ws + OFF_Y);

  // OLA accumulator must be zero every launch (graph replays included).
  hipMemsetAsync(Y, 0, SZ_Y, stream);

  const int nX = kRows * kDim;
  k_cast_x <<<(nX + 255) / 256, 256, 0, stream>>>(x, XB, nX);
  k_prep_w <<<(kK2 * kDim + 255) / 256, 256, 0, stream>>>(W, WB);
  k_basis  <<<dim3((kK2 + 255) / 256, kNfft), 256, 0, stream>>>(B2T);
  k_zero_padA<<<(kRows + 255) / 256, 256, 0, stream>>>(A2);

  // GEMM1: 65536 rows (4 waves * 16 rows per block) x 33 frequency tiles.
  k_gemm1_act<<<dim3(kRows / 64, 33), 128, 0, stream>>>(XB, WB, bias, A2);

  // GEMM2: 4096*16-row frame tiles x 64 sample tiles (4 per block).
  k_gemm2_ola<<<dim3(kRows / 16, kNfft / 64), 128, 0, stream>>>(A2, B2T, Y);

  k_finalize<<<(kBatch * kOutLen + 255) / 256, 256, 0, stream>>>(Y, out);
}
